// SimpleFrameTokenSelector_9225589752447
// MI455X (gfx1250) — matrix operand order repaired and checked by reading, not compiled, not run on hardware
//
#include <hip/hip_runtime.h>
#include <hip/hip_bf16.h>
#include <math.h>
#include <stdint.h>

// ---------------- problem constants ----------------
#define B_    8
#define T_    64
#define N_    197
#define D_    768
#define HIDD  3072   // HID*D
#define KF    8
#define KT    49
#define NPAD  208    // 13 tiles of 16

// output layout (flat f32, concatenated in return order)
#define Z_ELEMS     (B_*KF*KT*D_)        // 2,408,448
#define FIDX_ELEMS  (B_*KF)              // 64
#define TIDX_ELEMS  (B_*KF*KT)           // 3,136
#define FMASK_ELEMS (B_*T_)              // 512
#define TMASK_ELEMS (B_*T_*N_)           // 100,864

// workspace byte offsets (all 16B aligned)
#define OFF_HLN    0u                        // bf16 [512,768]
#define OFF_VALID  786432u                   // u32 [512]
#define OFF_W1T    788480u                   // bf16 [3072,768]
#define OFF_SCORE  5507072u                  // f32 [512]
#define OFF_FIDX   5509120u                  // i32 [64]
#define OFF_TIDX   5509632u                  // i32 [3136]
#define OFF_FNB16  5522176u                  // bf16 [512,768]
#define OFF_G      6308608u                  // f32 [64,208,208]
#define OFF_XBF    17384192u                 // bf16 [64,208,768]
#define OFF_NORMS  37831424u                 // f32 [64,208]
#define OFF_VLD    37884672u                 // i32 [64,208]

typedef __attribute__((ext_vector_type(16))) __bf16 v16bf;
typedef __attribute__((ext_vector_type(8)))  float  v8f;
typedef __attribute__((ext_vector_type(4)))  unsigned int u32x4;

union FragU { u32x4 u[2]; v16bf v; };

// A-matrix 16x32 bf16 fragment from row-major storage (row = M, col = K)
// lane<16: K = k0+{0..7, 16..23}; lane>=16: K = k0+{8..15, 24..31}
__device__ __forceinline__ v16bf load_fragA(const __bf16* base, int row0, int k0, int stride) {
  const int lane = threadIdx.x & 31;
  const int half = lane >> 4;
  const int r    = lane & 15;
  const __bf16* p = base + (size_t)(row0 + r) * stride + (k0 + (half ? 8 : 0));
  FragU f;
  f.u[0] = *(const u32x4*)(p);
  f.u[1] = *(const u32x4*)(p + 16);
  return f.v;
}

// B-matrix 32x16 bf16 fragment, B[k][n] = base[(row0+n)*stride + k] (transposed source rows)
// lane<16: K = k0+{0..7, 16..23}; lane>=16: K = k0+{8..15, 24..31}
__device__ __forceinline__ v16bf load_fragB(const __bf16* base, int row0, int k0, int stride) {
  const int lane = threadIdx.x & 31;
  const int half = lane >> 4;
  const int r    = lane & 15;
  const __bf16* p = base + (size_t)(row0 + r) * stride + (k0 + 8 * half);
  FragU f;
  f.u[0] = *(const u32x4*)(p);
  f.u[1] = *(const u32x4*)(p + 16);
  return f.v;
}

__device__ __forceinline__ v8f wmma_bf16f32(v16bf a, v16bf b, v8f c) {
  return __builtin_amdgcn_wmma_f32_16x16x32_bf16(false, a, false, b, (short)0, c, false, false);
}

// CDNA5 async global->LDS copy (16B per lane), tracked by ASYNCcnt
__device__ __forceinline__ void async_g2l_b128(unsigned ldsOff, const void* gaddr) {
  asm volatile("global_load_async_to_lds_b128 %0, %1, off"
               :: "v"(ldsOff), "v"(gaddr) : "memory");
}
__device__ __forceinline__ void wait_async0() {
  asm volatile("s_wait_asynccnt 0" ::: "memory");
}

__device__ __forceinline__ float gelu_exact(float v) {
  return 0.5f * v * (1.0f + erff(v * 0.70710678118654752f));
}

// ---------------- K0: zero token_mask output ----------------
__global__ void k0_zero(float* __restrict__ tmask) {
  int i = blockIdx.x * 256 + threadIdx.x;
  if (i < TMASK_ELEMS) tmask[i] = 0.0f;
}

// ---------------- K1: frame_repr -> fnb16 (normalized bf16), hln (LayerNorm bf16), valid ----------------
__global__ void __launch_bounds__(256) k1_frame(
    const float* __restrict__ x, const float* __restrict__ mask,
    const float* __restrict__ gamma, const float* __restrict__ beta,
    __bf16* __restrict__ fnb16, __bf16* __restrict__ hln, unsigned int* __restrict__ valid) {
  __shared__ float mk[N_];
  __shared__ float red[256];
  __shared__ float s_denom, s_mu, s_rstd, s_rnorm;
  const int row = blockIdx.x;           // b*T + t
  const int tid = threadIdx.x;
  const float* xr = x + (size_t)row * N_ * D_;

  if (tid < N_) mk[tid] = mask[(size_t)row * N_ + tid];
  __syncthreads();

  // denom = sum(mask)
  red[tid] = (tid < N_) ? mk[tid] : 0.0f;
  __syncthreads();
  for (int s = 128; s > 0; s >>= 1) { if (tid < s) red[tid] += red[tid + s]; __syncthreads(); }
  if (tid == 0) { float d = red[0]; valid[row] = (d > 0.0f) ? 1u : 0u; s_denom = fmaxf(d, 1e-6f); }
  __syncthreads();
  const float invden = 1.0f / s_denom;

  // masked mean over tokens; thread handles d = tid, tid+256, tid+512
  float a0 = 0.f, a1 = 0.f, a2 = 0.f;
  for (int n = 0; n < N_; ++n) {
    const float* rp = xr + (size_t)n * D_;
    if (n + 12 < N_) __builtin_prefetch(rp + 12 * D_ + tid, 0, 1);  // global_prefetch_b8
    float m = mk[n];
    a0 += rp[tid] * m;
    a1 += rp[tid + 256] * m;
    a2 += rp[tid + 512] * m;
  }
  float fr0 = a0 * invden, fr1 = a1 * invden, fr2 = a2 * invden;

  // mean / sumsq reductions
  red[tid] = fr0 + fr1 + fr2; __syncthreads();
  for (int s = 128; s > 0; s >>= 1) { if (tid < s) red[tid] += red[tid + s]; __syncthreads(); }
  if (tid == 0) s_mu = red[0] / (float)D_;
  __syncthreads();
  red[tid] = fr0*fr0 + fr1*fr1 + fr2*fr2; __syncthreads();
  for (int s = 128; s > 0; s >>= 1) { if (tid < s) red[tid] += red[tid + s]; __syncthreads(); }
  if (tid == 0) {
    float S2 = red[0];
    float mu = s_mu;
    float var = S2 / (float)D_ - mu * mu;
    s_rstd  = rsqrtf(var + 1e-5f);
    s_rnorm = 1.0f / fmaxf(sqrtf(S2), 1e-12f);
  }
  __syncthreads();

  const float mu = s_mu, rstd = s_rstd, rnorm = s_rnorm;
  float frv[3] = { fr0, fr1, fr2 };
  for (int j = 0; j < 3; ++j) {
    int d = tid + j * 256;
    float h = (frv[j] - mu) * rstd * gamma[d] + beta[d];
    hln[(size_t)row * D_ + d]   = (__bf16)h;
    fnb16[(size_t)row * D_ + d] = (__bf16)(frv[j] * rnorm);
  }
}

// ---------------- K1b: W1 [768,3072] f32 -> W1t [3072,768] bf16 ----------------
__global__ void k1b_w1t(const float* __restrict__ W1, __bf16* __restrict__ W1t) {
  int i = blockIdx.x * 256 + threadIdx.x;   // over 3072*768, k fastest
  if (i >= HIDD * D_) return;
  int n = i / D_;
  int k = i - n * D_;
  W1t[i] = (__bf16)W1[(size_t)k * HIDD + n];
}

// ---------------- K2: score = (GELU(hln @ W1 + b1)) @ W2 + b2, bf16 WMMA ----------------
__global__ void __launch_bounds__(256) k2_score(
    const __bf16* __restrict__ hln, const __bf16* __restrict__ W1t,
    const float* __restrict__ b1, const float* __restrict__ W2,
    const float* __restrict__ b2, float* __restrict__ score) {
  __shared__ __align__(16) __bf16 Al[16 * D_];   // 24 KB A-tile
  __shared__ float wpart[8][16];
  const int tid = threadIdx.x;
  const int m0  = blockIdx.x * 16;

  { // async-stage 16 rows of hln into LDS: 24,576 B = 6 * (256 lanes * 16B)
    const unsigned aBase = (unsigned)(uintptr_t)Al;
    const char* asrc = (const char*)(hln + (size_t)m0 * D_);
    for (int it = 0; it < 6; ++it) {
      int i = tid + it * 256;
      async_g2l_b128(aBase + i * 16, asrc + (size_t)i * 16);
    }
  }
  wait_async0();
  __syncthreads();

  const int wid  = tid >> 5;
  const int lane = tid & 31;
  const int half = lane >> 4;
  const int r    = lane & 15;
  float rowAcc[8];
  for (int e = 0; e < 8; ++e) rowAcc[e] = 0.0f;

  for (int nt = wid; nt < HIDD / 16; nt += 8) {   // 24 N-tiles per wave, uniform
    const int n0 = nt * 16;
    v8f acc = {};
    for (int k0 = 0; k0 < D_; k0 += 32) {
      v16bf a = load_fragA(Al, 0, k0, D_);
      v16bf b = load_fragB(W1t, n0, k0, D_);
      acc = wmma_bf16f32(a, b, acc);
    }
    const int   col  = n0 + r;
    const float bias = b1[col];
    const float w2v  = W2[col];
    for (int e = 0; e < 8; ++e)
      rowAcc[e] += gelu_exact(acc[e] + bias) * w2v;   // per-lane partial over its columns
  }
  // one 16-lane reduction per row AFTER the tile loop (not inside it)
  for (int e = 0; e < 8; ++e) {
    float p = rowAcc[e];
    for (int off = 8; off > 0; off >>= 1) p += __shfl_xor(p, off, 16);
    rowAcc[e] = p;
  }
  if (r == 0)
    for (int e = 0; e < 8; ++e) wpart[wid][e + 8 * half] = rowAcc[e];
  __syncthreads();
  if (tid < 16) {
    float s = 0.0f;
    for (int w = 0; w < 8; ++w) s += wpart[w][tid];   // fixed order -> deterministic
    score[m0 + tid] = s + b2[0];
  }
}

// ---------------- K3: cosine sim (WMMA) + greedy facility location ----------------
__global__ void __launch_bounds__(256) k3_facility(
    const __bf16* __restrict__ fnb16, const float* __restrict__ score,
    const unsigned int* __restrict__ valid, int* __restrict__ fidx_ws,
    float* __restrict__ out_fidx, float* __restrict__ out_fmask) {
  extern __shared__ char sm3[];
  __bf16* fnb    = (__bf16*)sm3;                         // 64*768 bf16 = 98,304 B
  float*  sim    = (float*)(sm3 + 64 * D_ * 2);          // 64*65 f32 (padded) = 16,640 B
  float*  best   = sim + 64 * 65;
  float*  tot    = best + 64;
  float*  scoreL = tot + 64;
  int*    chosen = (int*)(scoreL + 64);
  int*    validL = chosen + 64;
  int*    sel    = validL + 64;                          // 8
  int*    jbp    = sel + 8;

  const int b   = blockIdx.x;
  const int tid = threadIdx.x;

  { // async-stage fn rows (already bf16): 98,304 B = 24 * 4096
    const unsigned fBase = (unsigned)(uintptr_t)fnb;
    const char* fsrc = (const char*)(fnb16 + (size_t)b * T_ * D_);
    for (int it = 0; it < 24; ++it) {
      int i = tid + it * 256;
      async_g2l_b128(fBase + i * 16, fsrc + (size_t)i * 16);
    }
  }
  if (tid < 64) {
    best[tid] = 0.0f; chosen[tid] = 0;
    scoreL[tid] = score[b * 64 + tid];
    validL[tid] = (int)valid[b * 64 + tid];
  }
  wait_async0();
  __syncthreads();

  // sim = fnb @ fnb^T  (16 tiles of 16x16, 2 per wave)
  const int wid = tid >> 5;
  for (int t = wid; t < 16; t += 8) {
    const int m0 = (t >> 2) * 16, n0 = (t & 3) * 16;
    v8f acc = {};
    for (int k0 = 0; k0 < D_; k0 += 32) {
      v16bf a  = load_fragA(fnb, m0, k0, D_);
      v16bf bb = load_fragB(fnb, n0, k0, D_);
      acc = wmma_bf16f32(a, bb, acc);
    }
    const int lane = tid & 31, half = lane >> 4, r = lane & 15;
    for (int e = 0; e < 8; ++e) sim[(m0 + e + 8 * half) * 65 + (n0 + r)] = acc[e];
  }
  __syncthreads();

  for (int k = 0; k < KF; ++k) {
    if (tid < 64) {
      float cg = 0.0f;
      for (int i = 0; i < 64; ++i) cg += fmaxf(best[i], sim[i * 65 + tid]);
      float tv = cg + 0.5f * scoreL[tid];   // -sum(best) is constant over j: argmax unchanged
      if (!validL[tid] || chosen[tid]) tv = -INFINITY;
      tot[tid] = tv;
    }
    __syncthreads();
    if (tid == 0) {
      int jb = 0; float bv = tot[0];
      for (int j = 1; j < 64; ++j) if (tot[j] > bv) { bv = tot[j]; jb = j; }
      *jbp = jb; chosen[jb] = 1; sel[k] = jb;
    }
    __syncthreads();
    const int jb = *jbp;
    if (tid < 64) best[tid] = fmaxf(best[tid], sim[tid * 65 + jb]);
    __syncthreads();
  }

  if (tid < KF) { fidx_ws[b * KF + tid] = sel[tid]; out_fidx[b * KF + tid] = (float)sel[tid]; }
  if (tid < 64) out_fmask[b * 64 + tid] = chosen[tid] ? 1.0f : 0.0f;
}

// ---------------- K3b: gather selected frames -> bf16 tiles + exact f32 norms + valid ----------------
__global__ void __launch_bounds__(256) k3b_stage(
    const float* __restrict__ x, const float* __restrict__ mask,
    const int* __restrict__ fidx_ws, __bf16* __restrict__ Xbf,
    float* __restrict__ normsW, int* __restrict__ vldW) {
  const int blk = blockIdx.x;       // b*8 + kf
  const int b = blk >> 3;
  const int tid = threadIdx.x;
  const int fi = fidx_ws[blk];
  const float* xr = x + (size_t)(b * T_ + fi) * N_ * D_;
  __bf16* dst = Xbf + (size_t)blk * NPAD * D_;

  for (int i = tid; i < NPAD * D_; i += 256) {
    int n = i / D_;
    float v = (n < N_) ? xr[(size_t)n * D_ + (i - n * D_)] : 0.0f;
    dst[i] = (__bf16)v;
  }
  if (tid < NPAD) {
    float s = 0.0f;
    if (tid < N_) {
      const float* rp = xr + (size_t)tid * D_;
      for (int d = 0; d < D_; ++d) { float v = rp[d]; s += v * v; }
    }
    normsW[blk * NPAD + tid] = s;
    vldW[blk * NPAD + tid] = (tid < N_) && (mask[(size_t)(b * T_ + fi) * N_ + tid] > 0.5f);
  }
}

// ---------------- K4: Gram via WMMA (X async-staged into 312KB LDS) + greedy k-center + gathers ----------------
__global__ void __launch_bounds__(256) k4_kcenter(
    const float* __restrict__ x, const __bf16* __restrict__ Xbf,
    const float* __restrict__ normsW, const int* __restrict__ vldW,
    const int* __restrict__ fidx_ws, float* __restrict__ G,
    int* __restrict__ tidx_ws, float* __restrict__ out_z,
    float* __restrict__ out_tidx, float* __restrict__ out_tmask) {
  extern __shared__ char sm4[];
  __bf16* Xl     = (__bf16*)sm4;                         // NPAD*768 bf16 = 319,488 B
  float*  norms  = (float*)(sm4 + NPAD * D_ * 2);        // 832 B
  float*  md     = norms + NPAD;                         // 832 B
  int*    vldA   = (int*)(md + NPAD);                    // 832 B
  int*    tokSel = vldA + NPAD;                          // 196 B
  int*    cbp    = tokSel + KT;

  const int blk = blockIdx.x;       // b*8 + kf
  const int b = blk >> 3;
  const int tid = threadIdx.x;
  const int fi = fidx_ws[blk];
  const float* xr = x + (size_t)(b * T_ + fi) * N_ * D_;
  float* Gb = G + (size_t)blk * NPAD * NPAD;

  { // async-stage bf16 X tile: 319,488 B = 78 * 4096
    const unsigned xBase = (unsigned)(uintptr_t)Xl;
    const char* src = (const char*)(Xbf + (size_t)blk * NPAD * D_);
    for (int it = 0; it < 78; ++it) {
      int i = tid + it * 256;
      async_g2l_b128(xBase + i * 16, src + (size_t)i * 16);
    }
  }
  if (tid < NPAD) {
    norms[tid] = normsW[blk * NPAD + tid];
    vldA[tid]  = vldW[blk * NPAD + tid];
  }
  wait_async0();
  __syncthreads();

  // Gram G = X @ X^T  (13x13 tiles) via bf16 WMMA, f32 accumulate
  const int wid = tid >> 5, lane = tid & 31, half = lane >> 4, r = lane & 15;
  for (int t = wid; t < 169; t += 8) {     // wave-uniform
    const int m0 = (t / 13) * 16, n0 = (t % 13) * 16;
    v8f acc = {};
    for (int k0 = 0; k0 < D_; k0 += 32) {
      v16bf a  = load_fragA(Xl, m0, k0, D_);
      v16bf bb = load_fragB(Xl, n0, k0, D_);
      acc = wmma_bf16f32(a, bb, acc);
    }
    for (int e = 0; e < 8; ++e) Gb[(size_t)(m0 + e + 8 * half) * NPAD + (n0 + r)] = acc[e];
  }
  __threadfence();
  __syncthreads();

  // k-center: first = argmax ||x||^2 over valid
  if (tid == 0) {
    int first = 0; float bv = -INFINITY;
    for (int i = 0; i < N_; ++i) {
      float d0 = vldA[i] ? norms[i] : -INFINITY;
      if (d0 > bv) { bv = d0; first = i; }
    }
    tokSel[0] = first; *cbp = first;
  }
  __syncthreads();
  int cand = *cbp;
  if (tid < NPAD) {
    float v = -1.0f;
    if (vldA[tid]) {
      float d2 = norms[tid] + norms[cand] - 2.0f * Gb[(size_t)cand * NPAD + tid];
      v = sqrtf(fmaxf(d2, 0.0f));
    }
    md[tid] = v;
  }
  __syncthreads();
  for (int k = 1; k < KT; ++k) {
    if (tid == 0) {
      int c = 0; float bv = md[0];
      for (int i = 1; i < N_; ++i) if (md[i] > bv) { bv = md[i]; c = i; }
      tokSel[k] = c; *cbp = c;
    }
    __syncthreads();
    cand = *cbp;
    if (tid < NPAD && vldA[tid]) {
      float d2 = norms[tid] + norms[cand] - 2.0f * Gb[(size_t)cand * NPAD + tid];
      md[tid] = fminf(md[tid], sqrtf(fmaxf(d2, 0.0f)));
    }
    __syncthreads();
  }

  // outputs
  if (tid < KT) {
    int tk = tokSel[tid];
    tidx_ws[blk * KT + tid] = tk;
    out_tidx[blk * KT + tid] = (float)tk;
    out_tmask[(size_t)(b * T_ + fi) * N_ + tk] = 1.0f;
  }
  for (int i = tid; i < KT * D_; i += 256) {
    int kt = i / D_, d = i - kt * D_;
    out_z[(size_t)blk * KT * D_ + i] = xr[(size_t)tokSel[kt] * D_ + d];
  }
}

// ---------------- launch ----------------
extern "C" void kernel_launch(void* const* d_in, const int* in_sizes, int n_in,
                              void* d_out, int out_size, void* d_ws, size_t ws_size,
                              hipStream_t stream) {
  const float* x     = (const float*)d_in[0];
  const float* mask  = (const float*)d_in[1];
  const float* gamma = (const float*)d_in[2];
  const float* beta  = (const float*)d_in[3];
  const float* W1    = (const float*)d_in[4];
  const float* b1    = (const float*)d_in[5];
  const float* W2    = (const float*)d_in[6];
  const float* b2    = (const float*)d_in[7];

  char* ws = (char*)d_ws;
  __bf16*       hln   = (__bf16*)(ws + OFF_HLN);
  unsigned int* valid = (unsigned int*)(ws + OFF_VALID);
  __bf16*       W1t   = (__bf16*)(ws + OFF_W1T);
  float*        score = (float*)(ws + OFF_SCORE);
  int*          fidx  = (int*)(ws + OFF_FIDX);
  int*          tidx  = (int*)(ws + OFF_TIDX);
  __bf16*       fnb16 = (__bf16*)(ws + OFF_FNB16);
  float*        G     = (float*)(ws + OFF_G);
  __bf16*       Xbf   = (__bf16*)(ws + OFF_XBF);
  float*        normsW= (float*)(ws + OFF_NORMS);
  int*          vldW  = (int*)(ws + OFF_VLD);

  float* out       = (float*)d_out;
  float* out_z     = out;
  float* out_fidx  = out_z + Z_ELEMS;
  float* out_tidx  = out_fidx + FIDX_ELEMS;
  float* out_fmask = out_tidx + TIDX_ELEMS;
  float* out_tmask = out_fmask + FMASK_ELEMS;

  const size_t smem3 = (size_t)64 * D_ * 2 + 64 * 65 * 4 + 4 * 256 + 2 * 256 + 64; // ~116.3 KB
  const size_t smem4 = (size_t)NPAD * D_ * 2 + 3 * NPAD * 4 + (KT + 8) * 4;        // ~322.2 KB

  k0_zero    <<<(TMASK_ELEMS + 255) / 256, 256, 0, stream>>>(out_tmask);
  k1_frame   <<<B_ * T_, 256, 0, stream>>>(x, mask, gamma, beta, fnb16, hln, valid);
  k1b_w1t    <<<(HIDD * D_) / 256, 256, 0, stream>>>(W1, W1t);
  k2_score   <<<(B_ * T_) / 16, 256, 0, stream>>>(hln, W1t, b1, W2, b2, score);
  k3_facility<<<B_, 256, smem3, stream>>>(fnb16, score, valid, fidx, out_fidx, out_fmask);
  k3b_stage  <<<B_ * KF, 256, 0, stream>>>(x, mask, fidx, Xbf, normsW, vldW);
  k4_kcenter <<<B_ * KF, 256, smem4, stream>>>(x, Xbf, normsW, vldW, fidx, G, tidx,
                                               out_z, out_tidx, out_tmask);
}
